// ABCNN1_4286377362200
// MI455X (gfx1250) — matrix operand order repaired and checked by reading, not compiled, not run on hardware
//
#include <hip/hip_runtime.h>
#include <hip/hip_bf16.h>

#define B_ 32
#define S_ 512
#define D_ 768
#define O_ 256
#define KC_ (6 * D_)   // conv GEMM K extent

typedef __attribute__((ext_vector_type(16))) __bf16 v16bf;
typedef __attribute__((ext_vector_type(8)))  float  v8f;

union FragU {
    uint4  q[2];
    v16bf  v;
    __bf16 h[16];
};

// A fragment (16x32 bf16, M x K). Lane<16: M=lane, K={0..7,16..23}.
// Lane>=16: M=lane-16, K={8..15,24..31}. As is row-major [*, 32].
__device__ __forceinline__ v16bf ld_afrag(const __bf16* As, int row, int half) {
    FragU f;
    const __bf16* p = As + row * 32 + half * 8;
    f.q[0] = *(const uint4*)p;          // K = half*8 .. +7
    f.q[1] = *(const uint4*)(p + 16);   // K = half*8+16 .. +23
    return f.v;
}

// B fragment (32x16 bf16, K x N). Lane holds column N=lane%16, K={0..7,16..23}
// for lane<16, K={8..15,24..31} for lane>=16. Bs staged as [N][32].
__device__ __forceinline__ v16bf ld_bfrag(const __bf16* Bs, int col, int half) {
    FragU f;
    const __bf16* p = Bs + col * 32 + half * 8;
    f.q[0] = *(const uint4*)p;
    f.q[1] = *(const uint4*)(p + 16);
    return f.v;
}

// ---------------- prep kernels ----------------

__global__ void prep_rows_kernel(const float* __restrict__ xa, const float* __restrict__ xb,
                                 __bf16* __restrict__ xab, __bf16* __restrict__ xbb,
                                 float* __restrict__ na, float* __restrict__ nb) {
    int r = blockIdx.x;
    bool second = (blockIdx.y != 0);
    const float* src = second ? xb : xa;
    __bf16* dst = second ? xbb : xab;
    float* nrm = second ? nb : na;
    int tid = threadIdx.x;
    float s = 0.f;
    size_t base = (size_t)r * D_;
    for (int c = tid; c < D_; c += 256) {
        float v = src[base + c];
        dst[base + c] = (__bf16)v;
        s += v * v;
    }
    __shared__ float red[256];
    red[tid] = s;
    __syncthreads();
    for (int off = 128; off > 0; off >>= 1) {
        if (tid < off) red[tid] += red[tid + off];
        __syncthreads();
    }
    if (tid == 0) nrm[r] = red[0];
}

// weight [S=K][D=N] -> wbT [N=D][K=S] row-major bf16 (vectorizable B staging)
__global__ void prep_weight_kernel(const float* __restrict__ w, __bf16* __restrict__ wbT) {
    const int total = S_ * D_;
    int i = blockIdx.x * blockDim.x + threadIdx.x;
    int stride = gridDim.x * blockDim.x;
    for (; i < total; i += stride) {
        int n = i / S_;          // 0..D-1
        int k = i - n * S_;      // 0..S-1
        wbT[i] = (__bf16)w[(size_t)k * D_ + n];
    }
}

// conv_w [O,2,D,W] -> cwT [o][(ci*3+wi)*D + d]  (row-major N x K operand)
__global__ void prep_conv_kernel(const float* __restrict__ cw, __bf16* __restrict__ cwT) {
    const int total = O_ * KC_;
    int i = blockIdx.x * blockDim.x + threadIdx.x;
    int stride = gridDim.x * blockDim.x;
    for (; i < total; i += stride) {
        int o = i / KC_;
        int kk = i - o * KC_;
        int seg = kk / D_;
        int d = kk - seg * D_;
        int ci = seg / 3, wi = seg - ci * 3;
        cwT[i] = (__bf16)cw[(((size_t)o * 2 + ci) * D_ + d) * 3 + wi];
    }
}

// ---------------- shared GEMM micro-kernel pieces ----------------
// Block tile: 64 (M) x 256 (N), K step 32. 8 waves as 2 x 4, each 32x64.

#define BM_ 64
#define BN_ 256

struct Tile {
    int lane, half, lrow, wm, wn;
};

__device__ __forceinline__ Tile decode(int tid) {
    Tile t;
    t.lane = tid & 31;
    t.half = t.lane >> 4;
    t.lrow = t.lane & 15;
    int wid = tid >> 5;
    t.wm = wid >> 2;      // 0..1 -> 32 rows each
    t.wn = wid & 3;       // 0..3 -> 64 cols each
    return t;
}

__device__ __forceinline__ void mma_step(const __bf16* As, const __bf16* Bs,
                                         const Tile& t, v8f acc[2][4]) {
    v16bf af[2], bfv[4];
    af[0] = ld_afrag(As, t.wm * 32 + t.lrow,      t.half);
    af[1] = ld_afrag(As, t.wm * 32 + 16 + t.lrow, t.half);
#pragma unroll
    for (int j = 0; j < 4; ++j)
        bfv[j] = ld_bfrag(Bs, t.wn * 64 + j * 16 + t.lrow, t.half);
#pragma unroll
    for (int i = 0; i < 2; ++i)
#pragma unroll
        for (int j = 0; j < 4; ++j)
            acc[i][j] = __builtin_amdgcn_wmma_f32_16x16x32_bf16(
                false, af[i], false, bfv[j], (short)0, acc[i][j], false, false);
}

// stage a 256x32 B tile from [N][K] row-major source: one row per thread
__device__ __forceinline__ void stage_b(__bf16* Bs, const __bf16* src_nk,
                                        size_t ldk, int n0, int k0, int tid) {
    const uint4* s4 = (const uint4*)(src_nk + (size_t)(n0 + tid) * ldk + k0);
    uint4* d4 = (uint4*)&Bs[tid * 32];
    d4[0] = s4[0]; d4[1] = s4[1]; d4[2] = s4[2]; d4[3] = s4[3];
}

// ---------------- GEMM 1: attn = 1/(sqrt(max(na+nb-2*xa@xb^T,eps))+1) ----------------

__global__ void __launch_bounds__(256) attn_gemm_kernel(
        const __bf16* __restrict__ xab, const __bf16* __restrict__ xbb,
        const float* __restrict__ na, const float* __restrict__ nb,
        __bf16* __restrict__ Pa, __bf16* __restrict__ Pt) {
    __shared__ __bf16 As[BM_ * 32];
    __shared__ __bf16 Bs[BN_ * 32];
    const int tid = threadIdx.x;
    const int b = blockIdx.z;
    const int n0 = blockIdx.x * BN_;
    const int m0 = blockIdx.y * BM_;
    const __bf16* Abase = xab + (size_t)b * S_ * D_;
    const __bf16* Bbase = xbb + (size_t)b * S_ * D_;
    const Tile t = decode(tid);
    v8f acc[2][4] = {};

    for (int k0 = 0; k0 < D_; k0 += 32) {
        {   // stage A tile 64x32
            int m = tid >> 2, kk = (tid & 3) * 8;
            *(uint4*)&As[m * 32 + kk] =
                *(const uint4*)(Abase + (size_t)(m0 + m) * D_ + k0 + kk);
        }
        stage_b(Bs, Bbase, D_, n0, k0, tid);   // xb rows are [n][k] for B=xb^T
        __syncthreads();
        mma_step(As, Bs, t, acc);
        __syncthreads();
    }

    const size_t bSS = (size_t)b * S_ * S_;
    const int bS = b * S_;
#pragma unroll
    for (int i = 0; i < 2; ++i)
#pragma unroll
        for (int j = 0; j < 4; ++j) {
            int n = n0 + t.wn * 64 + j * 16 + t.lrow;
            float nbv = nb[bS + n];
#pragma unroll
            for (int g = 0; g < 8; ++g) {
                int m = m0 + t.wm * 32 + i * 16 + t.half * 8 + g;
                float d2 = na[bS + m] + nbv - 2.0f * acc[i][j][g];
                d2 = fmaxf(d2, 1e-12f);
                float at = 1.0f / (sqrtf(d2) + 1.0f);
                __bf16 h = (__bf16)at;
                Pa[bSS + (size_t)m * S_ + n] = h;
                Pt[bSS + (size_t)n * S_ + m] = h;
            }
        }
}

// ---------------- GEMM 2: xa_attn = Pa @ W ; xb_attn = Pt @ W ----------------

__global__ void __launch_bounds__(256) pw_gemm_kernel(
        const __bf16* __restrict__ Pa, const __bf16* __restrict__ Pt,
        const __bf16* __restrict__ wbT,
        __bf16* __restrict__ Aatt, __bf16* __restrict__ Batt) {
    __shared__ __bf16 As[BM_ * 32];
    __shared__ __bf16 Bs[BN_ * 32];
    const int tid = threadIdx.x;
    const int z = blockIdx.z;
    const int b = z >> 1, side = z & 1;
    const __bf16* P = (side ? Pt : Pa) + (size_t)b * S_ * S_;
    __bf16* Y = (side ? Batt : Aatt) + (size_t)b * S_ * D_;
    const int n0 = blockIdx.x * BN_;
    const int m0 = blockIdx.y * BM_;
    const Tile t = decode(tid);
    v8f acc[2][4] = {};

    for (int k0 = 0; k0 < S_; k0 += 32) {
        {
            int m = tid >> 2, kk = (tid & 3) * 8;
            *(uint4*)&As[m * 32 + kk] =
                *(const uint4*)(P + (size_t)(m0 + m) * S_ + k0 + kk);
        }
        stage_b(Bs, wbT, S_, n0, k0, tid);     // wbT is [D][S] = [N][K]
        __syncthreads();
        mma_step(As, Bs, t, acc);
        __syncthreads();
    }

#pragma unroll
    for (int i = 0; i < 2; ++i)
#pragma unroll
        for (int j = 0; j < 4; ++j) {
            int n = n0 + t.wn * 64 + j * 16 + t.lrow;
#pragma unroll
            for (int g = 0; g < 8; ++g) {
                int m = m0 + t.wm * 32 + i * 16 + t.half * 8 + g;
                Y[(size_t)m * D_ + n] = (__bf16)acc[i][j][g];
            }
        }
}

// ---------------- GEMM 3: conv-as-GEMM, K = 6 segments of D with row shifts ----------------

__global__ void __launch_bounds__(256) conv_gemm_kernel(
        const __bf16* __restrict__ xab, const __bf16* __restrict__ xbb,
        const __bf16* __restrict__ Aatt, const __bf16* __restrict__ Batt,
        const __bf16* __restrict__ cwT, const float* __restrict__ cb,
        float* __restrict__ out) {
    __shared__ __bf16 As[BM_ * 32];
    __shared__ __bf16 Bs[BN_ * 32];
    const int tid = threadIdx.x;
    const int z = blockIdx.z;
    const int b = z >> 1, side = z & 1;
    const __bf16* X0 = (side ? xbb : xab) + (size_t)b * S_ * D_;
    const __bf16* X1 = (side ? Batt : Aatt) + (size_t)b * S_ * D_;
    float* outp = out + (size_t)side * B_ * O_ * S_ + (size_t)b * O_ * S_;
    const int n0 = blockIdx.x * BN_;
    const int m0 = blockIdx.y * BM_;
    const Tile t = decode(tid);
    v8f acc[2][4] = {};

    for (int k0 = 0; k0 < KC_; k0 += 32) {
        int seg = k0 / D_;                 // constant within a 32-chunk (768 % 32 == 0)
        int ci = seg / 3;
        int shift = (seg - ci * 3) - 1;    // w - 1 in {-1,0,1}
        int d0 = k0 - seg * D_;
        const __bf16* src = ci ? X1 : X0;
        {   // stage shifted A rows, zero-padded at sequence boundary
            int m = tid >> 2, kk = (tid & 3) * 8;
            int row = m0 + m + shift;
            uint4 v = make_uint4(0u, 0u, 0u, 0u);
            if (row >= 0 && row < S_)
                v = *(const uint4*)(src + (size_t)row * D_ + d0 + kk);
            *(uint4*)&As[m * 32 + kk] = v;
        }
        stage_b(Bs, cwT, KC_, n0, k0, tid);    // cwT is [O][6D] = [N][K]
        __syncthreads();
        mma_step(As, Bs, t, acc);
        __syncthreads();
    }

#pragma unroll
    for (int i = 0; i < 2; ++i)
#pragma unroll
        for (int j = 0; j < 4; ++j) {
            int o = n0 + t.wn * 64 + j * 16 + t.lrow;
            float bias = cb[o];
#pragma unroll
            for (int g = 0; g < 8; ++g) {
                int s = m0 + t.wm * 32 + i * 16 + t.half * 8 + g;
                float v = acc[i][j][g] + bias;
                outp[(size_t)o * S_ + s] = fmaxf(v, 0.0f);
            }
        }
}

// ---------------- host ----------------

extern "C" void kernel_launch(void* const* d_in, const int* in_sizes, int n_in,
                              void* d_out, int out_size, void* d_ws, size_t ws_size,
                              hipStream_t stream) {
    (void)in_sizes; (void)n_in; (void)out_size; (void)ws_size;
    const float* xa  = (const float*)d_in[0];
    const float* xb  = (const float*)d_in[1];
    const float* wgt = (const float*)d_in[2];
    const float* cw  = (const float*)d_in[3];
    const float* cb  = (const float*)d_in[4];
    float* out = (float*)d_out;

    char* p = (char*)d_ws;
    auto take = [&](size_t bytes) { char* r = p; p += (bytes + 255) & ~(size_t)255; return r; };
    __bf16* xab  = (__bf16*)take((size_t)B_ * S_ * D_ * 2);
    __bf16* xbb  = (__bf16*)take((size_t)B_ * S_ * D_ * 2);
    __bf16* wbT  = (__bf16*)take((size_t)S_ * D_ * 2);
    __bf16* cwT  = (__bf16*)take((size_t)O_ * KC_ * 2);
    float*  na   = (float*) take((size_t)B_ * S_ * 4);
    float*  nb   = (float*) take((size_t)B_ * S_ * 4);
    __bf16* Pa   = (__bf16*)take((size_t)B_ * S_ * S_ * 2);
    __bf16* Pt   = (__bf16*)take((size_t)B_ * S_ * S_ * 2);
    __bf16* Aatt = (__bf16*)take((size_t)B_ * S_ * D_ * 2);
    __bf16* Batt = (__bf16*)take((size_t)B_ * S_ * D_ * 2);

    prep_rows_kernel<<<dim3(B_ * S_, 2), 256, 0, stream>>>(xa, xb, xab, xbb, na, nb);
    prep_weight_kernel<<<dim3(384), 256, 0, stream>>>(wgt, wbT);
    prep_conv_kernel<<<dim3(512), 256, 0, stream>>>(cw, cwT);

    attn_gemm_kernel<<<dim3(S_ / BN_, S_ / BM_, B_), 256, 0, stream>>>(xab, xbb, na, nb, Pa, Pt);
    pw_gemm_kernel<<<dim3(D_ / BN_, S_ / BM_, B_ * 2), 256, 0, stream>>>(Pa, Pt, wbT, Aatt, Batt);
    conv_gemm_kernel<<<dim3(O_ / BN_, S_ / BM_, B_ * 2), 256, 0, stream>>>(xab, xbb, Aatt, Batt, cwT, cb, out);
}
